// SequenceEBM_9328668967425
// MI455X (gfx1250) — matrix operand order repaired and checked by reading, not compiled, not run on hardware
//
#include <hip/hip_runtime.h>

// ---------------------------------------------------------------- constants
static constexpr int   NRES  = 16384;
static constexpr int   KNBR  = 30;
static constexpr long  NEDGE = (long)NRES * KNBR;     // 491520
static constexpr int   HID   = 128;

// LDS row strides (bf16 elements), padded to avoid bank conflicts
static constexpr int XS = 136;   // 128 + 8
static constexpr int AS = 328;   // 320 + 8
static constexpr int YS = 136;

// ---------------------------------------------------------------- ws layout (bytes)
static constexpr size_t OFF_SCALES = 0;                         // 12 f32
static constexpr size_t OFF_EW     = 256;                       // 7 * 32768 B packed edge weights
static constexpr size_t OFF_WROW   = OFF_EW + 7 * 32768;        // 128 f32 (=229632)
static constexpr size_t OFF_OW0    = OFF_WROW + 512;            // 20*120*512 bf16
static constexpr size_t OFF_OW1    = OFF_OW0 + 2457600;         // 8*10*512 bf16
static constexpr size_t OFF_OW2    = OFF_OW1 + 81920;           // 8*4*512 bf16
static constexpr size_t OFF_OW3    = OFF_OW2 + 32768;           // 2*4*512 bf16
static constexpr size_t OFF_SUMS   = OFF_OW3 + 8192;            // 64 f32
static constexpr size_t OFF_CNTS   = OFF_SUMS + 256;            // 64 f32
static constexpr size_t OFF_CAT    = OFF_CNTS + 256;            // NEDGE*128 bf16 (126 MB)

// ---------------------------------------------------------------- WMMA types
typedef __attribute__((ext_vector_type(16))) __bf16 v16bf;
typedef __attribute__((ext_vector_type(8)))  __bf16 bf16x8;
typedef __attribute__((ext_vector_type(8)))  float  v8f;

union AV { v16bf v; bf16x8 h[2]; };

__device__ __forceinline__ v8f wmma_bf16(const AV& a, const AV& b, v8f c) {
  return __builtin_amdgcn_wmma_f32_16x16x32_bf16(false, a.v, false, b.v,
                                                 (short)0, c, false, false);
}

// Generic dense layer over one 16-row M-subtile handled by one wave.
// Input rows at inRow (wave base already applied), weights pre-packed into
// B-fragments of 512 bf16 (frag index = nt*KT + kt, within-frag = lane*16+j).
template<int NT, int KT, bool RELU>
__device__ __forceinline__ void dense_layer(const __bf16* inRow, int inStride,
                                            const __bf16* __restrict__ wfr,
                                            const float*  __restrict__ bias,
                                            __bf16* outRow, int outStride,
                                            int lane)
{
  const int lm = lane & 15, hi = lane >> 4;
  const __bf16* ar = inRow + lm * inStride;
  AV a[KT];
#pragma unroll
  for (int kt = 0; kt < KT; ++kt) {
    const int kb = kt * 32 + hi * 8;   // A layout: lane<16 -> K {0..7,16..23}
    a[kt].h[0] = *(const bf16x8*)(ar + kb);
    a[kt].h[1] = *(const bf16x8*)(ar + kb + 16);
  }
#pragma unroll
  for (int nt = 0; nt < NT; ++nt) {
    v8f acc = {0.f,0.f,0.f,0.f,0.f,0.f,0.f,0.f};
#pragma unroll
    for (int kt = 0; kt < KT; ++kt) {
      AV b;
      const __bf16* bp = wfr + (size_t)(nt * KT + kt) * 512 + lane * 16;
      b.h[0] = *(const bf16x8*)bp;
      b.h[1] = *(const bf16x8*)(bp + 8);
      acc = wmma_bf16(a[kt], b, acc);
    }
    const int   col = nt * 16 + lm;
    const float bv  = bias[col];
#pragma unroll
    for (int r = 0; r < 8; ++r) {       // C layout: vgpr r -> row hi*8+r, col=lm
      float x = acc[r] + bv;
      if (RELU) x = fmaxf(x, 0.0f);
      outRow[(hi * 8 + r) * outStride + col] = (__bf16)x;
    }
  }
}

// ---------------------------------------------------------------- features
__device__ __forceinline__ void feat49(const float* __restrict__ primary,
                                       const float* __restrict__ angles,
                                       const float* __restrict__ orient,
                                       int n, int j, float* f)
{
  const float* on = orient + (size_t)n * 7;
  const float* oj = orient + (size_t)j * 7;
  const float dx = oj[0] - on[0], dy = oj[1] - on[1], dz = oj[2] - on[2];
  const float dist = sqrtf(dx*dx + dy*dy + dz*dz);

  float qnw = on[3], qnx = on[4], qny = on[5], qnz = on[6];
  float rn = sqrtf(qnw*qnw + qnx*qnx + qny*qny + qnz*qnz) + 1e-12f;
  qnw /= rn; qnx /= rn; qny /= rn; qnz /= rn;
  float qjw = oj[3], qjx = oj[4], qjy = oj[5], qjz = oj[6];
  float rj = sqrtf(qjw*qjw + qjx*qjx + qjy*qjy + qjz*qjz) + 1e-12f;
  qjw /= rj; qjx /= rj; qjy /= rj; qjz /= rj;

  for (int i = 0; i < 20; ++i) f[i]      = primary[(size_t)j * 20 + i];
  for (int i = 0; i < 6;  ++i) f[20 + i] = angles[(size_t)j * 6 + i];
#pragma unroll
  for (int i = 0; i < 16; ++i) {                 // MU = linspace(0,20,16), SIG=1.25
    const float t = (dist - (20.0f / 15.0f) * (float)i) * (1.0f / 1.25f);
    f[26 + i] = expf(-t * t);
  }
  const float inv = 1.0f / (dist + 1e-6f);
  const float ux = dx * inv, uy = dy * inv, uz = dz * inv;
  const float xx = qnx*qnx, yy = qny*qny, zz = qnz*qnz;
  const float xy = qnx*qny, xz = qnx*qnz, yz = qny*qnz;
  const float wx = qnw*qnx, wy = qnw*qny, wz = qnw*qnz;
  const float R00 = 1 - 2*(yy+zz), R01 = 2*(xy-wz), R02 = 2*(xz+wy);
  const float R10 = 2*(xy+wz), R11 = 1 - 2*(xx+zz), R12 = 2*(yz-wx);
  const float R20 = 2*(xz-wy), R21 = 2*(yz+wx), R22 = 1 - 2*(xx+yy);
  f[42] = R00*ux + R10*uy + R20*uz;              // R^T @ d_unit
  f[43] = R01*ux + R11*uy + R21*uz;
  f[44] = R02*ux + R12*uy + R22*uz;
  const float w1 = qnw, x1 = -qnx, y1 = -qny, z1 = -qnz;   // q_conj * q_j
  f[45] = w1*qjw - x1*qjx - y1*qjy - z1*qjz;
  f[46] = w1*qjx + qjw*x1 + (y1*qjz - z1*qjy);
  f[47] = w1*qjy + qjw*y1 + (z1*qjx - x1*qjz);
  f[48] = w1*qjz + qjw*z1 + (x1*qjy - y1*qjx);
}

// ---------------------------------------------------------------- spectral norm
__global__ __launch_bounds__(256)
void spectral_scale_kernel(const float* __restrict__ W, int R, int C,
                           float* __restrict__ outScale)
{
  __shared__ float u[384];
  __shared__ float v[3840];
  __shared__ float red[256];
  const int t = threadIdx.x;
  const float ini = rsqrtf((float)R);
  for (int r = t; r < R; r += 256) u[r] = ini;
  __syncthreads();
  float sigma = 1.0f;
  for (int it = 0; it < 5; ++it) {
    for (int c = t; c < C; c += 256) {           // v = W^T u
      float s = 0.f;
      for (int r = 0; r < R; ++r) s += W[(size_t)r * C + c] * u[r];
      v[c] = s;
    }
    __syncthreads();
    float p = 0.f;
    for (int c = t; c < C; c += 256) p += v[c] * v[c];
    red[t] = p; __syncthreads();
    for (int o = 128; o > 0; o >>= 1) { if (t < o) red[t] += red[t + o]; __syncthreads(); }
    const float nv = sqrtf(red[0]) + 1e-12f;
    __syncthreads();
    for (int c = t; c < C; c += 256) v[c] /= nv;
    __syncthreads();
    for (int r = t; r < R; r += 256) {           // u = W v
      float s = 0.f;
      for (int c = 0; c < C; ++c) s += W[(size_t)r * C + c] * v[c];
      u[r] = s;
    }
    __syncthreads();
    p = 0.f;
    for (int r = t; r < R; r += 256) p += u[r] * u[r];
    red[t] = p; __syncthreads();
    for (int o = 128; o > 0; o >>= 1) { if (t < o) red[t] += red[t + o]; __syncthreads(); }
    const float nu = sqrtf(red[0]);
    sigma = nu;                                   // sigma = u.(W v) = ||W v||
    __syncthreads();
    const float d = nu + 1e-12f;
    for (int r = t; r < R; r += 256) u[r] /= d;
    __syncthreads();
  }
  if (t == 0) outScale[0] = 1.0f / fmaxf(sigma, 1e-30f);
}

// ---------------------------------------------------------------- weight packing
__global__ void pack_weights_kernel(const float* __restrict__ W,
                                    const float* __restrict__ scalePtr,
                                    int R, int C, int ntiles, int ktiles,
                                    __bf16* __restrict__ dst)
{
  const int total = ntiles * ktiles * 512;
  const int idx = blockIdx.x * 256 + threadIdx.x;
  if (idx >= total) return;
  const float s = scalePtr[0];
  const int within = idx & 511, frag = idx >> 9;
  const int lane = within >> 4, j = within & 15;
  const int nt = frag / ktiles, kt = frag - nt * ktiles;
  const int n = nt * 16 + (lane & 15);
  const int k = kt * 32 + ((lane >= 16) ? 8 : 0) + (j & 7) + ((j >= 8) ? 16 : 0);
  float v = 0.0f;
  if (n < R && k < C) v = W[(size_t)n * C + k] * s;
  dst[idx] = (__bf16)v;
}

__global__ void pack_wrow_kernel(const float* __restrict__ W,
                                 const float* __restrict__ scalePtr,
                                 float* __restrict__ wrow)
{
  const int c = threadIdx.x;
  if (c < 128) wrow[c] = W[c] * scalePtr[0];
}

// ---------------------------------------------------------------- edge MLP
struct EdgeP {
  const float* primary; const float* angles; const float* orient; const int* conn;
  const __bf16 *fW0, *fW1, *fW2, *fW3, *wW0, *wW1, *wW2;
  const float  *fb0, *fb1, *fb2, *fb3, *wb0, *wb1, *wb2;
  const float* wrow; const float* wb3;
  __bf16* cat;
};

__global__ __launch_bounds__(128) void edge_mlp_kernel(EdgeP P)
{
  __shared__ __bf16 Xs[64 * XS];
  __shared__ __bf16 Ys[64 * XS];
  __shared__ __bf16 Zs[64 * XS];
  __shared__ float  wv[64];
  const int  tid  = threadIdx.x;
  const int  wave = tid >> 5, lane = tid & 31;
  const long e0   = (long)blockIdx.x * 64;

  // ---- build 98-feature rows directly in LDS (two threads per edge row)
  {
    const int  r = tid >> 1;
    const long e = e0 + r;
    const int  n  = (int)(e / KNBR);
    const int  kk = (int)(e - (long)n * KNBR);
    const int  j  = (tid & 1) ? P.conn[n * KNBR] : P.conn[n * KNBR + kk];
    float f[49];
    feat49(P.primary, P.angles, P.orient, n, j, f);
    const int cb = (tid & 1) ? 49 : 0;
    for (int i = 0; i < 49; ++i) Xs[r * XS + cb + i] = (__bf16)f[i];
    if (tid & 1) for (int c = 98; c < XS; ++c) Xs[r * XS + c] = (__bf16)0.0f;
  }
  __syncthreads();

  const int rb = wave * 16;
  // ---- gating MLP (98->128->128->128, relu) then scalar dot
  dense_layer<8, 4, true>(Xs + rb * XS, XS, P.wW0, P.wb0, Ys + rb * XS, XS, lane);
  __syncthreads();
  dense_layer<8, 4, true>(Ys + rb * XS, XS, P.wW1, P.wb1, Zs + rb * XS, XS, lane);
  __syncthreads();
  dense_layer<8, 4, true>(Zs + rb * XS, XS, P.wW2, P.wb2, Ys + rb * XS, XS, lane);
  __syncthreads();
  if (tid < 64) {
    float s = P.wb3[0];
    for (int c = 0; c < HID; ++c) s += (float)Ys[tid * XS + c] * P.wrow[c];
    wv[tid] = s;
  }
  __syncthreads();
  // ---- feature MLP (98->128->128->128->128), last layer fused with p*w store
  dense_layer<8, 4, true>(Xs + rb * XS, XS, P.fW0, P.fb0, Zs + rb * XS, XS, lane);
  __syncthreads();
  dense_layer<8, 4, true>(Zs + rb * XS, XS, P.fW1, P.fb1, Ys + rb * XS, XS, lane);
  __syncthreads();
  dense_layer<8, 4, true>(Ys + rb * XS, XS, P.fW2, P.fb2, Zs + rb * XS, XS, lane);
  __syncthreads();
  {
    const int lm = lane & 15, hi = lane >> 4;
    const __bf16* ar = Zs + (rb + lm) * XS;
    AV a[4];
#pragma unroll
    for (int kt = 0; kt < 4; ++kt) {
      const int kb = kt * 32 + hi * 8;
      a[kt].h[0] = *(const bf16x8*)(ar + kb);
      a[kt].h[1] = *(const bf16x8*)(ar + kb + 16);
    }
#pragma unroll
    for (int nt = 0; nt < 8; ++nt) {
      v8f acc = {0.f,0.f,0.f,0.f,0.f,0.f,0.f,0.f};
#pragma unroll
      for (int kt = 0; kt < 4; ++kt) {
        AV b;
        const __bf16* bp = P.fW3 + (size_t)(nt * 4 + kt) * 512 + lane * 16;
        b.h[0] = *(const bf16x8*)bp;
        b.h[1] = *(const bf16x8*)(bp + 8);
        acc = wmma_bf16(a[kt], b, acc);
      }
      const int   col = nt * 16 + lm;
      const float bv  = P.fb3[col];
#pragma unroll
      for (int r = 0; r < 8; ++r) {
        const int row = rb + hi * 8 + r;
        const float x = (acc[r] + bv) * wv[row];
        P.cat[(size_t)(e0 + row) * HID + col] = (__bf16)x;
      }
    }
  }
}

// ---------------------------------------------------------------- output MLP
struct OutP {
  const __bf16* cat; const float* primary; const int* pidx;
  const __bf16 *oW0, *oW1, *oW2, *oW3;
  const float  *ob0, *ob1, *ob2, *ob3;
  float* sums;
};

__global__ __launch_bounds__(128) void out_mlp_kernel(OutP P)
{
  __shared__ __bf16 A1[64 * AS];
  __shared__ __attribute__((aligned(16))) char smemB[64 * YS * 2];
  __bf16* A2 = (__bf16*)smemB;
  float*  EB = (float*)smemB;             // overlays A2 (dead after layer 2)
  const int tid = threadIdx.x, wave = tid >> 5, lane = tid & 31;
  const int lm = lane & 15, hi = lane >> 4;
  const int n0 = blockIdx.x * 64;
  const int rb = wave * 16;

  // ---- layer 0: cat (K=3840) -> A1 (N=320, cols>=300 zero-padded)
  const __bf16* ar0 = P.cat + (size_t)(n0 + rb + lm) * 3840;
  for (int half = 0; half < 2; ++half) {
    v8f acc[10];
#pragma unroll
    for (int t2 = 0; t2 < 10; ++t2) acc[t2] = (v8f){0.f,0.f,0.f,0.f,0.f,0.f,0.f,0.f};
    for (int kc = 0; kc < 120; ++kc) {
      AV a;
      const int kb = kc * 32 + hi * 8;
      a.h[0] = *(const bf16x8*)(ar0 + kb);
      a.h[1] = *(const bf16x8*)(ar0 + kb + 16);
      if (kc + 2 < 120) __builtin_prefetch(ar0 + kb + 64, 0, 1);
#pragma unroll
      for (int t2 = 0; t2 < 10; ++t2) {
        const int nt = half * 10 + t2;
        AV b;
        const __bf16* bp = P.oW0 + (size_t)(nt * 120 + kc) * 512 + lane * 16;
        b.h[0] = *(const bf16x8*)bp;
        b.h[1] = *(const bf16x8*)(bp + 8);
        acc[t2] = wmma_bf16(a, b, acc[t2]);
      }
    }
#pragma unroll
    for (int t2 = 0; t2 < 10; ++t2) {
      const int col = (half * 10 + t2) * 16 + lm;
#pragma unroll
      for (int r = 0; r < 8; ++r) {
        float x = 0.0f;
        if (col < 300) x = fmaxf(acc[t2][r] + P.ob0[col], 0.0f);
        A1[(rb + hi * 8 + r) * AS + col] = (__bf16)x;
      }
    }
  }
  __syncthreads();
  // ---- layer 1: A1 (K=320) -> A2 (N=128)
  dense_layer<8, 10, true>(A1 + rb * AS, AS, P.oW1, P.ob1, A2 + rb * YS, YS, lane);
  __syncthreads();
  // ---- layer 2: A2 -> front of A1 (YS stride)
  dense_layer<8, 4, true>(A2 + rb * YS, YS, P.oW2, P.ob2, A1 + rb * YS, YS, lane);
  __syncthreads();
  // ---- layer 3: A1 -> EB (N padded 20->32, no relu)
  {
    const __bf16* ar = A1 + (rb + lm) * YS;
    AV a[4];
#pragma unroll
    for (int kt = 0; kt < 4; ++kt) {
      const int kb = kt * 32 + hi * 8;
      a[kt].h[0] = *(const bf16x8*)(ar + kb);
      a[kt].h[1] = *(const bf16x8*)(ar + kb + 16);
    }
#pragma unroll
    for (int nt = 0; nt < 2; ++nt) {
      v8f acc = {0.f,0.f,0.f,0.f,0.f,0.f,0.f,0.f};
#pragma unroll
      for (int kt = 0; kt < 4; ++kt) {
        AV b;
        const __bf16* bp = P.oW3 + (size_t)(nt * 4 + kt) * 512 + lane * 16;
        b.h[0] = *(const bf16x8*)bp;
        b.h[1] = *(const bf16x8*)(bp + 8);
        acc = wmma_bf16(a[kt], b, acc);
      }
      const int col = nt * 16 + lm;
      if (col < 20) {
        const float bv = P.ob3[col];
#pragma unroll
        for (int r = 0; r < 8; ++r)
          EB[(rb + hi * 8 + r) * 32 + col] = acc[r] + bv;
      }
    }
  }
  __syncthreads();
  // ---- energy gather: out[n, argmax(primary[n])] -> protein sums
  if (tid < 64) {
    const int n = n0 + tid;
    const float* pr = P.primary + (size_t)n * 20;
    float best = pr[0]; int bi = 0;
    for (int c = 1; c < 20; ++c) { const float x = pr[c]; if (x > best) { best = x; bi = c; } }
    atomicAdd(&P.sums[P.pidx[n]], EB[tid * 32 + bi]);
  }
}

// ---------------------------------------------------------------- tail kernels
__global__ void count_kernel(const int* __restrict__ pidx, float* __restrict__ cnts)
{
  const int i = blockIdx.x * 256 + threadIdx.x;
  if (i < NRES) atomicAdd(&cnts[pidx[i]], 1.0f);
}

__global__ void finalize_kernel(const float* __restrict__ sums,
                                const float* __restrict__ cnts,
                                float* __restrict__ out)
{
  const int p = threadIdx.x;
  if (p < 64) out[p] = sums[p] / fmaxf(cnts[p], 1.0f);
}

// ---------------------------------------------------------------- launch
extern "C" void kernel_launch(void* const* d_in, const int* in_sizes, int n_in,
                              void* d_out, int out_size, void* d_ws, size_t ws_size,
                              hipStream_t stream)
{
  (void)in_sizes; (void)n_in; (void)out_size; (void)ws_size;
  const float* primary = (const float*)d_in[0];
  const float* angles  = (const float*)d_in[2];
  const float* orient  = (const float*)d_in[3];
  const float* fW[4] = {(const float*)d_in[4],  (const float*)d_in[5],
                        (const float*)d_in[6],  (const float*)d_in[7]};
  const float* fb[4] = {(const float*)d_in[8],  (const float*)d_in[9],
                        (const float*)d_in[10], (const float*)d_in[11]};
  const float* wW[4] = {(const float*)d_in[12], (const float*)d_in[13],
                        (const float*)d_in[14], (const float*)d_in[15]};
  const float* wb[4] = {(const float*)d_in[16], (const float*)d_in[17],
                        (const float*)d_in[18], (const float*)d_in[19]};
  const float* oW[4] = {(const float*)d_in[20], (const float*)d_in[21],
                        (const float*)d_in[22], (const float*)d_in[23]};
  const float* ob[4] = {(const float*)d_in[24], (const float*)d_in[25],
                        (const float*)d_in[26], (const float*)d_in[27]};
  const int* conn = (const int*)d_in[28];
  const int* pidx = (const int*)d_in[29];

  char* W = (char*)d_ws;
  float*  scales = (float*)(W + OFF_SCALES);
  __bf16* ewp[7];
  for (int i = 0; i < 7; ++i) ewp[i] = (__bf16*)(W + OFF_EW + (size_t)i * 32768);
  float*  wrow = (float*)(W + OFF_WROW);
  __bf16* oW0p = (__bf16*)(W + OFF_OW0);
  __bf16* oW1p = (__bf16*)(W + OFF_OW1);
  __bf16* oW2p = (__bf16*)(W + OFF_OW2);
  __bf16* oW3p = (__bf16*)(W + OFF_OW3);
  float*  sums = (float*)(W + OFF_SUMS);
  float*  cnts = (float*)(W + OFF_CNTS);
  __bf16* cat  = (__bf16*)(W + OFF_CAT);

  // --- spectral-norm scales (12 matrices)
  const float* sm[12] = {fW[0], fW[1], fW[2], fW[3], wW[0], wW[1], wW[2], wW[3],
                         oW[0], oW[1], oW[2], oW[3]};
  const int sR[12] = {128,128,128,128, 128,128,128,1,   300,128,128,20};
  const int sC[12] = {98, 128,128,128, 98, 128,128,128, 3840,300,128,128};
  for (int i = 0; i < 12; ++i)
    spectral_scale_kernel<<<1, 256, 0, stream>>>(sm[i], sR[i], sC[i], scales + i);

  // --- pack weights into WMMA B-fragment layout
  auto pack = [&](const float* src, int slot, int R, int C, int nt, int kt, __bf16* dst) {
    const int total = nt * kt * 512;
    pack_weights_kernel<<<(total + 255) / 256, 256, 0, stream>>>(
        src, scales + slot, R, C, nt, kt, dst);
  };
  pack(fW[0], 0, 128, 98,  8, 4, ewp[0]);
  pack(fW[1], 1, 128, 128, 8, 4, ewp[1]);
  pack(fW[2], 2, 128, 128, 8, 4, ewp[2]);
  pack(fW[3], 3, 128, 128, 8, 4, ewp[3]);
  pack(wW[0], 4, 128, 98,  8, 4, ewp[4]);
  pack(wW[1], 5, 128, 128, 8, 4, ewp[5]);
  pack(wW[2], 6, 128, 128, 8, 4, ewp[6]);
  pack_wrow_kernel<<<1, 128, 0, stream>>>(wW[3], scales + 7, wrow);
  pack(oW[0], 8, 300, 3840, 20, 120, oW0p);
  pack(oW[1], 9, 128, 300,  8,  10,  oW1p);
  pack(oW[2], 10, 128, 128, 8,  4,   oW2p);
  pack(oW[3], 11, 20,  128, 2,  4,   oW3p);

  // --- edge MLP (features + gated feature outputs -> cat)
  EdgeP ep;
  ep.primary = primary; ep.angles = angles; ep.orient = orient; ep.conn = conn;
  ep.fW0 = ewp[0]; ep.fW1 = ewp[1]; ep.fW2 = ewp[2]; ep.fW3 = ewp[3];
  ep.wW0 = ewp[4]; ep.wW1 = ewp[5]; ep.wW2 = ewp[6];
  ep.fb0 = fb[0]; ep.fb1 = fb[1]; ep.fb2 = fb[2]; ep.fb3 = fb[3];
  ep.wb0 = wb[0]; ep.wb1 = wb[1]; ep.wb2 = wb[2];
  ep.wrow = wrow; ep.wb3 = wb[3];
  ep.cat = cat;
  edge_mlp_kernel<<<(int)(NEDGE / 64), 128, 0, stream>>>(ep);

  // --- segment accumulators
  hipMemsetAsync(W + OFF_SUMS, 0, 512, stream);   // sums + cnts
  count_kernel<<<NRES / 256, 256, 0, stream>>>(pidx, cnts);

  // --- output MLP + energy gather
  OutP op;
  op.cat = cat; op.primary = primary; op.pidx = pidx;
  op.oW0 = oW0p; op.oW1 = oW1p; op.oW2 = oW2p; op.oW3 = oW3p;
  op.ob0 = ob[0]; op.ob1 = ob[1]; op.ob2 = ob[2]; op.ob3 = ob[3];
  op.sums = sums;
  out_mlp_kernel<<<NRES / 64, 128, 0, stream>>>(op);

  finalize_kernel<<<1, 64, 0, stream>>>(sums, cnts, (float*)d_out);
}
